// SupBarlowTwinsLoss_35003983463132
// MI455X (gfx1250) — matrix operand order repaired and checked by reading, not compiled, not run on hardware
//
#include <hip/hip_runtime.h>

#define NB 16384     // batch == num classes
#define FD 128       // feature dim
#define LAMBDA 0.005f
#define LSTRIDE 40   // LDS column stride (elements); mult of 8 -> 16B-aligned b128

typedef __attribute__((ext_vector_type(16))) __bf16 v16bf;
typedef __attribute__((ext_vector_type(8)))  __bf16 v8bf;
typedef __attribute__((ext_vector_type(8)))  float  v8f;

// ---------- helpers ----------

__device__ __forceinline__ unsigned short f2bf(float f) {
    unsigned u = __float_as_uint(f);
    u += 0x7FFFu + ((u >> 16) & 1u);   // round-to-nearest-even
    return (unsigned short)(u >> 16);
}

__device__ __forceinline__ float wave_sum(float s) {
#pragma unroll
    for (int o = 16; o; o >>= 1) s += __shfl_xor(s, o, 32);
    return s;
}

// two contiguous 16B LDS fragments -> one 16-element bf16 WMMA operand
__device__ __forceinline__ v16bf combine(const unsigned short* lo, const unsigned short* hi) {
    v8bf a = *(const v8bf*)lo;
    v8bf b = *(const v8bf*)hi;
    return __builtin_shufflevector(a, b, 0, 1, 2, 3, 4, 5, 6, 7,
                                         8, 9, 10, 11, 12, 13, 14, 15);
}

// ---------- kernels ----------

__global__ void zero_kernel(float* p, int n) {
    for (int i = blockIdx.x * blockDim.x + threadIdx.x; i < n; i += gridDim.x * blockDim.x)
        p[i] = 0.0f;
}

// segment-sum of z into sums[label] and counts[label] via HW float atomics
__global__ __launch_bounds__(256) void scatter_kernel(const float* __restrict__ z,
                                                      const int* __restrict__ labels,
                                                      float* __restrict__ sums,
                                                      float* __restrict__ counts) {
    int idx = blockIdx.x * 256 + threadIdx.x;      // NB*32 threads, 4 floats each
    int row = idx >> 5;
    int ch  = idx & 31;
    int lab = labels[row];
    float4 v = ((const float4*)(z + row * FD))[ch];
    float* d = sums + lab * FD + ch * 4;
    unsafeAtomicAdd(d + 0, v.x);
    unsafeAtomicAdd(d + 1, v.y);
    unsafeAtomicAdd(d + 2, v.z);
    unsafeAtomicAdd(d + 3, v.w);
    if (ch == 0) unsafeAtomicAdd(&counts[lab], 1.0f);
}

// one wave per row: normalize z-row and center-row, exact fp32 diagonal terms,
// and emit bf16 copies TRANSPOSED (column-major [FD][NB]) via an LDS tile so
// global stores remain b128-vectorized.
__global__ __launch_bounds__(256) void normalize_kernel(const float* __restrict__ z,
                                                        const float* __restrict__ sums,
                                                        const float* __restrict__ counts,
                                                        unsigned short* __restrict__ znT,
                                                        unsigned short* __restrict__ cnT,
                                                        float* __restrict__ scal) {
    __shared__ unsigned short ztile[FD * 8];   // col-major: [col]*8 + wave
    __shared__ unsigned short ctile[FD * 8];

    int wv   = threadIdx.x >> 5;
    int lane = threadIdx.x & 31;
    int r0   = blockIdx.x * 8;
    int row  = r0 + wv;

    float4 z4 = ((const float4*)(z + row * FD))[lane];
    float zs = wave_sum(z4.x * z4.x + z4.y * z4.y + z4.z * z4.z + z4.w * z4.w);
    float zr = 1.0f / fmaxf(sqrtf(zs), 1e-12f);
    float4 zn = make_float4(z4.x * zr, z4.y * zr, z4.z * zr, z4.w * zr);

    float cnt = counts[row];
    float ic  = cnt > 0.0f ? 1.0f / cnt : 0.0f;
    float4 c4 = ((const float4*)(sums + row * FD))[lane];
    c4 = make_float4(c4.x * ic, c4.y * ic, c4.z * ic, c4.w * ic);
    float cs = wave_sum(c4.x * c4.x + c4.y * c4.y + c4.z * c4.z + c4.w * c4.w);
    float cr = 1.0f / fmaxf(sqrtf(cs), 1e-12f);
    float4 cn = make_float4(c4.x * cr, c4.y * cr, c4.z * cr, c4.w * cr);

    int c0 = lane * 4;
    ztile[(c0 + 0) * 8 + wv] = f2bf(zn.x);
    ztile[(c0 + 1) * 8 + wv] = f2bf(zn.y);
    ztile[(c0 + 2) * 8 + wv] = f2bf(zn.z);
    ztile[(c0 + 3) * 8 + wv] = f2bf(zn.w);
    ctile[(c0 + 0) * 8 + wv] = f2bf(cn.x);
    ctile[(c0 + 1) * 8 + wv] = f2bf(cn.y);
    ctile[(c0 + 2) * 8 + wv] = f2bf(cn.z);
    ctile[(c0 + 3) * 8 + wv] = f2bf(cn.w);

    float d = wave_sum(zn.x * cn.x + zn.y * cn.y + zn.z * cn.z + zn.w * cn.w);
    if (lane == 0) {
        d *= (1.0f / (float)NB);
        unsafeAtomicAdd(&scal[0], (d - 1.0f) * (d - 1.0f));  // invariance
        unsafeAtomicAdd(&scal[1], d * d);                    // diag squared
    }

    __syncthreads();
    int sel = threadIdx.x >> 7;          // 0 = zn, 1 = cn
    int col = threadIdx.x & 127;
    const unsigned short* t = (sel ? ctile : ztile) + col * 8;
    uint4 v = *(const uint4*)t;          // 8 bf16 = column of the 8-row tile
    unsigned short* dstT = sel ? cnT : znT;
    *(uint4*)(dstT + col * NB + r0) = v; // transposed, 16B-aligned (r0 % 8 == 0)
}

// G = M^T M via v_wmma_f32_16x16x32_bf16. Inputs are column-major [FD][NB] bf16,
// so a 32-row K-tile stages into column-major LDS with pure b128 copies and
// every WMMA operand fragment is a contiguous ds_load_b128.
// Software-pipelined: next tile prefetched into registers during compute.
// sched_barrier splits load/compute phases so all 18 ds_load_b128 issue as one
// deep clause and WMMAs retire against partial s_wait_dscnt values.
// blockIdx.y selects Zn vs Cn, blockIdx.x a 2048-row K-slice; atomics merge.
__global__ __launch_bounds__(256) void gram_kernel(const unsigned short* __restrict__ mat0T,
                                                   const unsigned short* __restrict__ mat1T,
                                                   float* __restrict__ G0,
                                                   float* __restrict__ G1) {
    const unsigned short* matT = blockIdx.y ? mat1T : mat0T;
    float* G                   = blockIdx.y ? G1 : G0;

    __shared__ unsigned short tile[FD * LSTRIDE];   // col-major, padded stride

    int tid  = threadIdx.x;
    int w    = tid >> 5;        // wave 0..7 -> output row-block
    int lane = tid & 31;
    int half = lane >> 4;
    int l    = lane & 15;

    int sc = tid >> 2;          // staging column 0..63 (and +64)
    int sp = (tid & 3) * 8;     // staging 8-element part within the 32-row column

    v8f acc[8] = {};            // 8 output tiles (16x16 f32) per wave

    int rowbase = blockIdx.x * 2048;

    // prologue: prefetch first 32 x 128 bf16 tile into registers
    uint4 r0 = *(const uint4*)(matT + sc * NB + rowbase + sp);
    uint4 r1 = *(const uint4*)(matT + (sc + 64) * NB + rowbase + sp);

    for (int ks = 0; ks < 2048; ks += 32) {
        // commit staged registers to LDS (waits on prefetch issued last iter)
        *(uint4*)(tile + sc * LSTRIDE + sp)        = r0;
        *(uint4*)(tile + (sc + 64) * LSTRIDE + sp) = r1;
        __syncthreads();

        // prefetch NEXT tile; latency overlaps with the compute phase below
        if (ks + 32 < 2048) {
            int k1 = rowbase + ks + 32;
            r0 = *(const uint4*)(matT + sc * NB + k1 + sp);
            r1 = *(const uint4*)(matT + (sc + 64) * NB + k1 + sp);
        }

        // ---- load phase: A + all 8 B operands (18 x ds_load_b128) ----
        // A operand: col = 16w + l ; frags at kk = {8h..8h+7, 16+8h..23+8h}
        const unsigned short* ac = tile + ((w << 4) + l) * LSTRIDE;
        v16bf a = combine(ac + (half << 3), ac + 16 + (half << 3));

        v16bf b[8];
#pragma unroll
        for (int n = 0; n < 8; ++n) {
            // B operand: col = 16n + l ; frags at kk = {8h..8h+7, 16+8h..23+8h}
            const unsigned short* bc = tile + ((n << 4) + l) * LSTRIDE;
            b[n] = combine(bc + (half << 3), bc + 16 + (half << 3));
        }

        // fence scheduling: keep all B operands live, loads ahead of WMMAs
        __builtin_amdgcn_sched_barrier(0);

        // ---- compute phase: 8 WMMAs with partial in-order DS waits ----
#pragma unroll
        for (int n = 0; n < 8; ++n)
            acc[n] = __builtin_amdgcn_wmma_f32_16x16x32_bf16(
                false, a, false, b[n], (short)0, acc[n], false, false);

        __syncthreads();   // all reads done before next iteration overwrites
    }

    // C/D layout: lane t<16 -> N=t, M=r ; lane t>=16 -> N=t-16, M=r+8
    int rbase = (w << 4) + (half << 3);
#pragma unroll
    for (int n = 0; n < 8; ++n)
#pragma unroll
        for (int r = 0; r < 8; ++r)
            unsafeAtomicAdd(&G[(rbase + r) * FD + (n << 4) + l], acc[n][r]);
}

__global__ __launch_bounds__(256) void finalize_kernel(const float* __restrict__ Gz,
                                                       const float* __restrict__ Gc,
                                                       const float* __restrict__ scal,
                                                       float* __restrict__ out) {
    __shared__ float red[256];
    float s = 0.0f;
    for (int i = threadIdx.x; i < FD * FD; i += 256) s += Gz[i] * Gc[i];
    red[threadIdx.x] = s;
    __syncthreads();
#pragma unroll
    for (int o = 128; o; o >>= 1) {
        if (threadIdx.x < o) red[threadIdx.x] += red[threadIdx.x + o];
        __syncthreads();
    }
    if (threadIdx.x == 0) {
        float S = red[0] / ((float)NB * (float)NB);   // sum of c^2 (incl. diag)
        out[0] = scal[0] + LAMBDA * (S - scal[1]);
    }
}

// ---------- launcher ----------

extern "C" void kernel_launch(void* const* d_in, const int* in_sizes, int n_in,
                              void* d_out, int out_size, void* d_ws, size_t ws_size,
                              hipStream_t stream) {
    const float* z      = (const float*)d_in[0];
    const int*   labels = (const int*)d_in[1];
    float* out = (float*)d_out;

    char* ws = (char*)d_ws;
    float*          sums   = (float*)(ws);                      // 16384*128 f32
    float*          counts = (float*)(ws + 8388608);            // 16384 f32
    unsigned short* znT    = (unsigned short*)(ws + 8454144);   // [128][16384] bf16
    unsigned short* cnT    = (unsigned short*)(ws + 12648448);  // [128][16384] bf16
    float*          Gz     = (float*)(ws + 16842752);           // 128*128 f32
    float*          Gc     = (float*)(ws + 16908288);           // 128*128 f32
    float*          scal   = (float*)(ws + 16973824);           // 2 f32

    zero_kernel<<<1024, 256, 0, stream>>>(sums, NB * FD + NB);   // sums+counts (contiguous)
    zero_kernel<<<64, 256, 0, stream>>>(Gz, FD * FD * 2 + 2);    // Gz+Gc+scal (contiguous)

    scatter_kernel<<<NB * 32 / 256, 256, 0, stream>>>(z, labels, sums, counts);
    normalize_kernel<<<NB / 8, 256, 0, stream>>>(z, sums, counts, znT, cnT, scal);

    dim3 gg(8, 2);
    gram_kernel<<<gg, 256, 0, stream>>>(znT, cnT, Gz, Gc);

    finalize_kernel<<<1, 256, 0, stream>>>(Gz, Gc, scal, out);
}
